// WideDeep_MD_adp_6227702579417
// MI455X (gfx1250) — compile-verified, matrix operand order checked by reading, not run on hardware
//
#include <hip/hip_runtime.h>

// ---------------------------------------------------------------------------
// Problem constants (match reference)
// ---------------------------------------------------------------------------
#define BATCH   16384
#define NW      8
#define ND      16
#define EMBD    16
#define VOCAB   100000
#define WIDEF   128      // NW*EMBD
#define DEEPF   256      // ND*EMBD
#define HIN     384      // WIDEF+DEEPF
#define KADP    32
#define EPSBN   1e-5f

// ---------------------------------------------------------------------------
// Types for WMMA (gfx1250: v_wmma_f32_16x16x32_bf16)
// ---------------------------------------------------------------------------
typedef __attribute__((ext_vector_type(16))) __bf16       bf16x16;
typedef __attribute__((ext_vector_type(8)))  float        f32x8;
typedef __attribute__((ext_vector_type(4)))  unsigned int u32x4;

union Frag16 { bf16x16 v; u32x4 q[2]; };

static __device__ __forceinline__ unsigned short f2bf(float f) {
  unsigned int u = __float_as_uint(f);
  u += 0x7FFFu + ((u >> 16) & 1u);           // round-to-nearest-even
  return (unsigned short)(u >> 16);
}

// ---------------------------------------------------------------------------
// Weight pack: f32 W[K,N] -> bf16 Wt[N,K]   (one-time, tiny)
// ---------------------------------------------------------------------------
__global__ void pack_w_bf16t(const float* __restrict__ W,
                             unsigned short* __restrict__ Wt, int K, int N) {
  int idx = blockIdx.x * blockDim.x + threadIdx.x;
  if (idx >= K * N) return;
  int k = idx / N, n = idx % N;
  Wt[(size_t)n * K + k] = f2bf(W[idx]);
}

__global__ void pack_f32t(const float* __restrict__ W,
                          float* __restrict__ Wt, int K, int N) {
  int idx = blockIdx.x * blockDim.x + threadIdx.x;
  if (idx >= K * N) return;
  int k = idx / N, n = idx % N;
  Wt[(size_t)n * K + k] = W[idx];
}

__global__ void zero_f32(float* __restrict__ p, int n) {
  int i = blockIdx.x * blockDim.x + threadIdx.x;
  if (i < n) p[i] = 0.f;
}

// ---------------------------------------------------------------------------
// Embedding gather:  Xh bf16[B,384] (wide||deep), Xd bf16[B,256], y_wide f32[B]
// ---------------------------------------------------------------------------
__global__ __launch_bounds__(128)
void gather_kernel(const int* __restrict__ wide_idx, const int* __restrict__ deep_idx,
                   const float* __restrict__ emb_wide, const float* __restrict__ emb_deep,
                   const float* __restrict__ lr_w, const float* __restrict__ lr_b,
                   unsigned short* __restrict__ Xh, unsigned short* __restrict__ Xd,
                   float* __restrict__ y_wide) {
  int b = blockIdx.x;
  int t = threadIdx.x;                        // 0..127
  int f = t >> 4, d = t & 15;
  int wi = wide_idx[b * NW + f];
  float wv = emb_wide[((size_t)f * VOCAB + wi) * EMBD + d];
  Xh[(size_t)b * HIN + t] = f2bf(wv);
  for (int j = t; j < DEEPF; j += 128) {
    int df = j >> 4, dd = j & 15;
    int di = deep_idx[b * ND + df];
    float dv = emb_deep[((size_t)df * VOCAB + di) * EMBD + dd];
    unsigned short bv = f2bf(dv);
    Xh[(size_t)b * HIN + WIDEF + j] = bv;
    Xd[(size_t)b * DEEPF + j] = bv;
  }
  __shared__ float red[128];
  red[t] = wv * lr_w[t];
  __syncthreads();
  for (int s = 64; s > 0; s >>= 1) {
    if (t < s) red[t] += red[t + s];
    __syncthreads();
  }
  if (t == 0) y_wide[b] = red[0] + lr_b[0];
}

// ---------------------------------------------------------------------------
// WMMA GEMM:  Y[B,N] = A[B,K](bf16) @ Wt[N,K](bf16)^T + bias, plus column stats.
// Block: 256 thr = 8 waves; wave tile 32x32 (4 accum frags); block tile 128x64.
// K-step 32: per lane 8x global_load_b128 feeding 4x v_wmma_f32_16x16x32_bf16.
// ---------------------------------------------------------------------------
__global__ __launch_bounds__(256)
void gemm_bf16_stats(const unsigned short* __restrict__ A,
                     const unsigned short* __restrict__ Wt,
                     const float* __restrict__ bias,
                     float* __restrict__ Y,
                     float* __restrict__ colsum, float* __restrict__ colsumsq,
                     int K, int N) {
  const int lane = threadIdx.x & 31;
  const int wave = threadIdx.x >> 5;
  const int wm = wave >> 1;                   // 0..3
  const int wn = wave & 1;                    // 0..1
  const int row0 = blockIdx.x * 128 + wm * 32;
  const int col0 = blockIdx.y * 64 + wn * 32;
  const int l16 = lane & 15;
  const int lh = lane >> 4;                   // lane half (0/1)

  f32x8 acc[2][2];
  acc[0][0] = f32x8{}; acc[0][1] = f32x8{}; acc[1][0] = f32x8{}; acc[1][1] = f32x8{};

  const unsigned short* a0p = A + (size_t)(row0 + l16) * K;
  const unsigned short* a1p = A + (size_t)(row0 + 16 + l16) * K;
  const unsigned short* b0p = Wt + (size_t)(col0 + l16) * K;
  const unsigned short* b1p = Wt + (size_t)(col0 + 16 + l16) * K;
  const int off0 = lh * 8;                    // 16-bit A/B frag: elems 0..7 at k+h*8
  const int off1 = 16 + lh * 8;               //                  elems 8..15 at k+16+h*8

  for (int k = 0; k < K; k += 32) {
    Frag16 a0, a1, b0, b1;
    a0.q[0] = *(const u32x4*)(a0p + k + off0);
    a0.q[1] = *(const u32x4*)(a0p + k + off1);
    a1.q[0] = *(const u32x4*)(a1p + k + off0);
    a1.q[1] = *(const u32x4*)(a1p + k + off1);
    b0.q[0] = *(const u32x4*)(b0p + k + off0);
    b0.q[1] = *(const u32x4*)(b0p + k + off1);
    b1.q[0] = *(const u32x4*)(b1p + k + off0);
    b1.q[1] = *(const u32x4*)(b1p + k + off1);
    if (k + 32 < K) {
      __builtin_prefetch(a0p + k + 32 + off0);
      __builtin_prefetch(b0p + k + 32 + off0);
    }
    acc[0][0] = __builtin_amdgcn_wmma_f32_16x16x32_bf16(false, a0.v, false, b0.v,
                                                        (short)0, acc[0][0], false, false);
    acc[0][1] = __builtin_amdgcn_wmma_f32_16x16x32_bf16(false, a0.v, false, b1.v,
                                                        (short)0, acc[0][1], false, false);
    acc[1][0] = __builtin_amdgcn_wmma_f32_16x16x32_bf16(false, a1.v, false, b0.v,
                                                        (short)0, acc[1][0], false, false);
    acc[1][1] = __builtin_amdgcn_wmma_f32_16x16x32_bf16(false, a1.v, false, b1.v,
                                                        (short)0, acc[1][1], false, false);
  }

  // Epilogue: C/D layout -> lane l16 = column, vgpr v = row (v + 8*lh) in tile.
  for (int nj = 0; nj < 2; nj++) {
    int n = col0 + nj * 16 + l16;
    float bcol = bias ? bias[n] : 0.f;
    float s = 0.f, s2 = 0.f;
    for (int mi = 0; mi < 2; mi++) {
      int rbase = row0 + mi * 16 + lh * 8;
      #pragma unroll
      for (int v = 0; v < 8; v++) {
        float val = acc[mi][nj][v] + bcol;
        Y[(size_t)(rbase + v) * N + n] = val;
        s += val; s2 += val * val;
      }
    }
    if (colsum) {
      atomicAdd(&colsum[n], s);
      atomicAdd(&colsumsq[n], s2);
    }
  }
}

// ---------------------------------------------------------------------------
// BN finalize: scale = g/sqrt(var+eps); shift = beta - mean*scale
// varDiv = 1/B (biased) or 1/(B-1) (adapter, ddof=1)
// ---------------------------------------------------------------------------
__global__ void bn_finalize(const float* __restrict__ colsum, const float* __restrict__ colsumsq,
                            const float* __restrict__ g, const float* __restrict__ beta,
                            float* __restrict__ scale, float* __restrict__ shift,
                            int N, float invCount, float varDiv) {
  int n = blockIdx.x * blockDim.x + threadIdx.x;
  if (n >= N) return;
  float mean = colsum[n] * invCount;
  float var = (colsumsq[n] - colsum[n] * mean) * varDiv;
  float sc = g[n] * rsqrtf(var + EPSBN);
  scale[n] = sc;
  shift[n] = beta[n] - mean * sc;
}

// Apply BN (+ReLU), write bf16 (next GEMM input) and/or f32 (may alias Y)
__global__ void bn_apply(const float* __restrict__ Y,
                         const float* __restrict__ scale, const float* __restrict__ shift,
                         unsigned short* __restrict__ out_bf16, float* __restrict__ out_f32,
                         int N, long total) {
  long stride = (long)gridDim.x * blockDim.x;
  for (long i = (long)blockIdx.x * blockDim.x + threadIdx.x; i < total; i += stride) {
    int n = (int)(i % N);
    float v = fmaxf(Y[i] * scale[n] + shift[n], 0.f);
    if (out_bf16) out_bf16[i] = f2bf(v);
    if (out_f32)  out_f32[i] = v;
  }
}

// ---------------------------------------------------------------------------
// Adapter: one wave32 per sample. Per-sample 32x32 mixing matrix from Hbuf.
//   t = sigmoid( ((yd@u0) x H[b]) @ v0 + b0 )
//   T2 = ((t@u1) x H[b]) @ v1 + b1         -> T2[B,256] + unbiased BN stats
// ---------------------------------------------------------------------------
__global__ __launch_bounds__(256)
void adapter_kernel(const float* __restrict__ yd, const float* __restrict__ u0t,
                    const float* __restrict__ H, const float* __restrict__ v0,
                    const float* __restrict__ u1, const float* __restrict__ v1,
                    const float* __restrict__ b0, const float* __restrict__ b1,
                    float* __restrict__ T2,
                    float* __restrict__ colsum, float* __restrict__ colsumsq) {
  const int lane = threadIdx.x & 31;
  const int b = blockIdx.x * 8 + (threadIdx.x >> 5);
  const float* ydr = yd + (size_t)b * DEEPF;
  const float* Hb = H + (size_t)b * (KADP * KADP);

  // t0[lane] = sum_f yd[f] * u0t[lane*256+f]
  float t0 = 0.f;
  const float* u0r = u0t + lane * DEEPF;
  for (int f = 0; f < DEEPF; f++) t0 += ydr[f] * u0r[f];

  // s[j] = sum_i t0_i * H[i,j]
  float s = 0.f;
  for (int i = 0; i < KADP; i++) s += __shfl(t0, i, 32) * Hb[i * KADP + lane];
  // r = sigmoid(s @ v0 + b0)
  float r = b0[lane];
  for (int j = 0; j < KADP; j++) r += __shfl(s, j, 32) * v0[j * KADP + lane];
  r = 1.f / (1.f + __expf(-r));
  // q = r @ u1
  float q = 0.f;
  for (int j = 0; j < KADP; j++) q += __shfl(r, j, 32) * u1[j * KADP + lane];
  // s2[j] = sum_i q_i * H[i,j]
  float s2 = 0.f;
  for (int i = 0; i < KADP; i++) s2 += __shfl(q, i, 32) * Hb[i * KADP + lane];
  // T2[n] = s2 @ v1 + b1, lane handles n = lane + 32*c
  float outv[8];
  #pragma unroll
  for (int c = 0; c < 8; c++) outv[c] = b1[lane + 32 * c];
  for (int j = 0; j < KADP; j++) {
    float sj = __shfl(s2, j, 32);
    const float* vr = v1 + j * DEEPF;
    #pragma unroll
    for (int c = 0; c < 8; c++) outv[c] += sj * vr[lane + 32 * c];
  }
  float* T2r = T2 + (size_t)b * DEEPF;
  #pragma unroll
  for (int c = 0; c < 8; c++) T2r[lane + 32 * c] = outv[c];

  // block-level BN stat reduction (8 samples/block) then global atomics
  __shared__ float ls[256], ls2[256];
  ls[threadIdx.x] = 0.f; ls2[threadIdx.x] = 0.f;
  __syncthreads();
  #pragma unroll
  for (int c = 0; c < 8; c++) {
    atomicAdd(&ls[lane + 32 * c], outv[c]);
    atomicAdd(&ls2[lane + 32 * c], outv[c] * outv[c]);
  }
  __syncthreads();
  atomicAdd(&colsum[threadIdx.x], ls[threadIdx.x]);
  atomicAdd(&colsumsq[threadIdx.x], ls2[threadIdx.x]);
}

// ---------------------------------------------------------------------------
// Final head: y = sigmoid(y_wide + (BN(T2) + yd) @ fw + fb)
// ---------------------------------------------------------------------------
__global__ __launch_bounds__(256)
void final_kernel(const float* __restrict__ T2,
                  const float* __restrict__ scale, const float* __restrict__ shift,
                  const float* __restrict__ yd, const float* __restrict__ fw,
                  const float* __restrict__ fb, const float* __restrict__ y_wide,
                  float* __restrict__ out) {
  const int lane = threadIdx.x & 31;
  const int b = blockIdx.x * 8 + (threadIdx.x >> 5);
  const float* t2r = T2 + (size_t)b * DEEPF;
  const float* ydr = yd + (size_t)b * DEEPF;
  float acc = 0.f;
  #pragma unroll
  for (int c = 0; c < 8; c++) {
    int n = lane + 32 * c;
    float o = t2r[n] * scale[n] + shift[n];
    acc += (o + ydr[n]) * fw[n];
  }
  for (int off = 16; off > 0; off >>= 1) acc += __shfl_down(acc, off, 32);
  if (lane == 0) out[b] = 1.f / (1.f + __expf(-(y_wide[b] + acc + fb[0])));
}

// ---------------------------------------------------------------------------
// Host launcher
// ---------------------------------------------------------------------------
extern "C" void kernel_launch(void* const* d_in, const int* in_sizes, int n_in,
                              void* d_out, int out_size, void* d_ws, size_t ws_size,
                              hipStream_t stream) {
  (void)in_sizes; (void)n_in; (void)out_size; (void)ws_size;
  const int*   wide_idx = (const int*)d_in[0];
  const int*   deep_idx = (const int*)d_in[1];
  // d_in[2] = domain_id (unused, identity selection)
  const float* emb_wide = (const float*)d_in[3];
  const float* emb_deep = (const float*)d_in[4];
  const float* lr_w = (const float*)d_in[5];
  const float* lr_b = (const float*)d_in[6];
  const float* mw1 = (const float*)d_in[7];   const float* mb1 = (const float*)d_in[8];
  const float* mg1 = (const float*)d_in[9];   const float* mbe1 = (const float*)d_in[10];
  const float* mw2 = (const float*)d_in[11];  const float* mb2 = (const float*)d_in[12];
  const float* mg2 = (const float*)d_in[13];  const float* mbe2 = (const float*)d_in[14];
  const float* hw1 = (const float*)d_in[15];  const float* hb1 = (const float*)d_in[16];
  const float* hg1 = (const float*)d_in[17];  const float* hbe1 = (const float*)d_in[18];
  const float* hw2 = (const float*)d_in[19];  const float* hb2 = (const float*)d_in[20];
  const float* hg2 = (const float*)d_in[21];  const float* hbe2 = (const float*)d_in[22];
  const float* hw3 = (const float*)d_in[23];  const float* hb3 = (const float*)d_in[24];
  const float* hg3 = (const float*)d_in[25];  const float* hbe3 = (const float*)d_in[26];
  const float* u0 = (const float*)d_in[27];   const float* u1 = (const float*)d_in[28];
  const float* v0 = (const float*)d_in[29];   const float* v1 = (const float*)d_in[30];
  const float* b0 = (const float*)d_in[31];   const float* b1 = (const float*)d_in[32];
  const float* gamma1 = (const float*)d_in[33];const float* bias1 = (const float*)d_in[34];
  const float* fw = (const float*)d_in[35];   const float* fb = (const float*)d_in[36];
  float* out = (float*)d_out;

  char* ws = (char*)d_ws;
  size_t off = 0;
  auto alloc = [&](size_t bytes) -> void* {
    off = (off + 255) & ~(size_t)255;
    void* p = ws + off;
    off += bytes;
    return p;
  };
  const size_t B = BATCH;
  unsigned short* Xh   = (unsigned short*)alloc(B * HIN * 2);
  unsigned short* Xd   = (unsigned short*)alloc(B * DEEPF * 2);
  unsigned short* A1   = (unsigned short*)alloc(B * 512 * 2);
  unsigned short* A2   = (unsigned short*)alloc(B * 256 * 2);
  unsigned short* B1   = (unsigned short*)alloc(B * 512 * 2);
  float* Hbuf  = (float*)alloc(B * 1024 * 4);
  float* Ydbuf = (float*)alloc(B * 256 * 4);
  float* Yscr  = (float*)alloc(B * 512 * 4);
  float* T2    = (float*)alloc(B * 256 * 4);
  float* ywide = (float*)alloc(B * 4);
  unsigned short* hw1t = (unsigned short*)alloc(384 * 512 * 2);
  unsigned short* hw2t = (unsigned short*)alloc(512 * 256 * 2);
  unsigned short* hw3t = (unsigned short*)alloc(256 * 1024 * 2);
  unsigned short* mw1t = (unsigned short*)alloc(256 * 512 * 2);
  unsigned short* mw2t = (unsigned short*)alloc(512 * 256 * 2);
  float* u0t   = (float*)alloc(32 * 256 * 4);
  float* colsum   = (float*)alloc(1024 * 4);
  float* colsumsq = (float*)alloc(1024 * 4);
  float* scale    = (float*)alloc(1024 * 4);
  float* shift    = (float*)alloc(1024 * 4);

  const float invB = 1.f / (float)B;
  const float invBm1 = 1.f / (float)(B - 1);

  auto packw = [&](const float* W, unsigned short* Wt, int K, int N) {
    int tot = K * N;
    pack_w_bf16t<<<(tot + 255) / 256, 256, 0, stream>>>(W, Wt, K, N);
  };
  packw(hw1, hw1t, 384, 512);
  packw(hw2, hw2t, 512, 256);
  packw(hw3, hw3t, 256, 1024);
  packw(mw1, mw1t, 256, 512);
  packw(mw2, mw2t, 512, 256);
  pack_f32t<<<(256 * 32 + 255) / 256, 256, 0, stream>>>(u0, u0t, 256, 32);

  gather_kernel<<<BATCH, 128, 0, stream>>>(wide_idx, deep_idx, emb_wide, emb_deep,
                                           lr_w, lr_b, Xh, Xd, ywide);

  auto zero_stats = [&]() { zero_f32<<<8, 256, 0, stream>>>(colsum, 1024);
                            zero_f32<<<8, 256, 0, stream>>>(colsumsq, 1024); };
  auto gemm = [&](const unsigned short* Aact, const unsigned short* Wt,
                  const float* bias, float* Y, int K, int N) {
    dim3 g(BATCH / 128, N / 64);
    gemm_bf16_stats<<<g, 256, 0, stream>>>(Aact, Wt, bias, Y, colsum, colsumsq, K, N);
  };
  auto bnfin = [&](const float* g, const float* be, int N, float varDiv) {
    bn_finalize<<<(N + 255) / 256, 256, 0, stream>>>(colsum, colsumsq, g, be,
                                                     scale, shift, N, invB, varDiv);
  };
  auto bnapp = [&](const float* Y, unsigned short* ob, float* of, int N) {
    bn_apply<<<2048, 256, 0, stream>>>(Y, scale, shift, ob, of, N, (long)B * N);
  };

  // ---- hyper net ----
  zero_stats(); gemm(Xh, hw1t, hb1, Yscr, 384, 512);
  bnfin(hg1, hbe1, 512, invB);  bnapp(Yscr, A1, nullptr, 512);
  zero_stats(); gemm(A1, hw2t, hb2, Yscr, 512, 256);
  bnfin(hg2, hbe2, 256, invB);  bnapp(Yscr, A2, nullptr, 256);
  zero_stats(); gemm(A2, hw3t, hb3, Hbuf, 256, 1024);
  bnfin(hg3, hbe3, 1024, invB); bnapp(Hbuf, nullptr, Hbuf, 1024);   // f32 in-place

  // ---- deep MLP ----
  zero_stats(); gemm(Xd, mw1t, mb1, Yscr, 256, 512);
  bnfin(mg1, mbe1, 512, invB);  bnapp(Yscr, B1, nullptr, 512);
  zero_stats(); gemm(B1, mw2t, mb2, Ydbuf, 512, 256);
  bnfin(mg2, mbe2, 256, invB);  bnapp(Ydbuf, nullptr, Ydbuf, 256);  // yd f32 in-place

  // ---- adapter (per-sample K x K mixing) ----
  zero_stats();
  adapter_kernel<<<BATCH / 8, 256, 0, stream>>>(Ydbuf, u0t, Hbuf, v0, u1, v1,
                                                b0, b1, T2, colsum, colsumsq);
  // adapter BN uses unbiased variance (ddof=1)
  bn_finalize<<<1, 256, 0, stream>>>(colsum, colsumsq, gamma1, bias1,
                                     scale, shift, 256, invB, invBm1);

  final_kernel<<<BATCH / 8, 256, 0, stream>>>(T2, scale, shift, Ydbuf, fw, fb,
                                              ywide, out);
}
